// MambaLayer_80985903333490
// MI455X (gfx1250) — matrix operand order repaired and checked by reading, not compiled, not run on hardware
//
#include <hip/hip_runtime.h>

typedef __attribute__((ext_vector_type(16))) __bf16 v16bf;
typedef __attribute__((ext_vector_type(8)))  float  v8f;

#define B_SZ   2
#define LSEQ   1024
#define DMODEL 1024
#define DSTATE 16
#define DCONV  4
#define DINNER 2048
#define NTOK   (B_SZ * LSEQ)      // 2048 tokens
#define DXZ    (2 * DINNER)       // 4096
#define NDBC   (2 * DSTATE + 1)   // 33
#define TCHUNK 64
#define NCHUNK (LSEQ / TCHUNK)    // 16

__device__ __forceinline__ float silu_f(float x)     { return x / (1.f + __expf(-x)); }
__device__ __forceinline__ float softplus_f(float x) { return (x > 20.f) ? x : log1pf(__expf(x)); }

// ---------------------------------------------------------------- LayerNorm -> bf16
__global__ void ln_kernel(const float* __restrict__ x, const float* __restrict__ g,
                          const float* __restrict__ b, __bf16* __restrict__ xn) {
  const int tok = blockIdx.x;
  const float* row = x + (size_t)tok * DMODEL;
  float s = 0.f, s2 = 0.f;
  for (int i = threadIdx.x; i < DMODEL; i += blockDim.x) { float v = row[i]; s += v; s2 += v * v; }
  __shared__ float red0[256], red1[256];
  red0[threadIdx.x] = s; red1[threadIdx.x] = s2;
  __syncthreads();
  for (int off = 128; off > 0; off >>= 1) {
    if (threadIdx.x < off) { red0[threadIdx.x] += red0[threadIdx.x + off];
                             red1[threadIdx.x] += red1[threadIdx.x + off]; }
    __syncthreads();
  }
  const float mu  = red0[0] * (1.f / DMODEL);
  const float var = red1[0] * (1.f / DMODEL) - mu * mu;
  const float rs  = rsqrtf(var + 1e-5f);
  for (int i = threadIdx.x; i < DMODEL; i += blockDim.x)
    xn[(size_t)tok * DMODEL + i] = (__bf16)((row[i] - mu) * rs * g[i] + b[i]);
}

// -------------------------------- one-time: W[K][N] fp32 -> Wt[N][K] bf16 (tiled)
__global__ void transpose_bf16_kernel(const float* __restrict__ W, __bf16* __restrict__ Wt,
                                      int K, int N) {
  __shared__ float tile[32][33];
  const int kt = blockIdx.y * 32, nt = blockIdx.x * 32;
  for (int r = threadIdx.y; r < 32; r += 8)
    tile[r][threadIdx.x] = W[(size_t)(kt + r) * N + nt + threadIdx.x];
  __syncthreads();
  for (int r = threadIdx.y; r < 32; r += 8)
    Wt[(size_t)(nt + r) * K + kt + threadIdx.x] = (__bf16)tile[threadIdx.x][r];
}

// ---------------- WMMA bf16 GEMM: C[M][N] = A[M][K] * Bt[N][K]^T (+resid)
// 256 threads = 8 waves (2x4). Block tile 128x128, K-chunks of 32.
// Double-buffered LDS, register-staged global prefetch, 1 barrier per chunk.
template <bool RESID>
__global__ void wmma_gemm_bf16(const __bf16* __restrict__ A, const __bf16* __restrict__ Bt,
                               const float* __restrict__ resid, float* __restrict__ C,
                               int M, int N, int K) {
  constexpr int BM = 128, BN = 128, KC = 32;
  __shared__ __align__(16) __bf16 sA[2][BM * KC];
  __shared__ __align__(16) __bf16 sB[2][BN * KC];
  const int tid  = threadIdx.x;
  const int lane = tid & 31;
  const int wave = tid >> 5;
  const int waveM = wave >> 2, waveN = wave & 3;   // 2 x 4 wave grid
  const int m0 = blockIdx.y * BM, n0 = blockIdx.x * BN;

  // staging: 512 groups of 8 bf16 per tile; thread owns groups {tid, tid+256}
  uint4 ra[2], rb[2];
  auto loadTiles = [&](int k0) {
#pragma unroll
    for (int i = 0; i < 2; ++i) {
      const int g = tid + i * 256;
      const int row = g >> 2, kg = g & 3;
      ra[i] = *(const uint4*)(A  + (size_t)(m0 + row) * K + k0 + kg * 8);
      rb[i] = *(const uint4*)(Bt + (size_t)(n0 + row) * K + k0 + kg * 8);
    }
  };
  auto storeTiles = [&](int buf) {
#pragma unroll
    for (int i = 0; i < 2; ++i) {
      const int g = tid + i * 256;
      const int row = g >> 2, kg = g & 3;
      const int swz = ((kg & 1) << 4) | ((kg >> 1) << 3);   // {0,16,8,24}: A-frag order
      *(uint4*)&sA[buf][row * KC + swz]    = ra[i];
      *(uint4*)&sB[buf][row * KC + swz]    = rb[i];         // same frag order as A
    }
  };

  v8f acc[4][2];
#pragma unroll
  for (int mi = 0; mi < 4; ++mi)
#pragma unroll
    for (int ni = 0; ni < 2; ++ni)
#pragma unroll
      for (int r = 0; r < 8; ++r) acc[mi][ni][r] = 0.f;

  const int nk = K / KC;
  loadTiles(0);
  storeTiles(0);
  for (int c = 0; c < nk; ++c) {
    __syncthreads();                       // waits dscnt: buf c&1 staged, buf (c+1)&1 free
    if (c + 1 < nk) loadTiles((c + 1) * KC);
    const __bf16* pa = &sA[c & 1][0];
    const __bf16* pb = &sB[c & 1][0];
    const int half16 = (lane >> 4) << 4;
    v16bf af[4], bfr[2];
#pragma unroll
    for (int mi = 0; mi < 4; ++mi)
      af[mi] = *(const v16bf*)&pa[(waveM * 64 + mi * 16 + (lane & 15)) * KC + half16];
#pragma unroll
    for (int ni = 0; ni < 2; ++ni)
      bfr[ni] = *(const v16bf*)&pb[(waveN * 32 + ni * 16 + (lane & 15)) * KC + half16];
#pragma unroll
    for (int mi = 0; mi < 4; ++mi)
#pragma unroll
      for (int ni = 0; ni < 2; ++ni)
        acc[mi][ni] = __builtin_amdgcn_wmma_f32_16x16x32_bf16(
            false, af[mi], false, bfr[ni], (short)0, acc[mi][ni], false, false);
    if (c + 1 < nk) storeTiles((c + 1) & 1);
  }

  // C/D layout: VGPR r -> M=r (lanes 0-15) / M=r+8 (lanes 16-31); N = lane%16.
#pragma unroll
  for (int mi = 0; mi < 4; ++mi)
#pragma unroll
    for (int ni = 0; ni < 2; ++ni)
#pragma unroll
      for (int r = 0; r < 8; ++r) {
        const int row = m0 + waveM * 64 + mi * 16 + r + ((lane >> 4) << 3);
        const int col = n0 + waveN * 32 + ni * 16 + (lane & 15);
        float v = acc[mi][ni][r];
        if (RESID) v += resid[(size_t)row * N + col];
        C[(size_t)row * N + col] = v;
      }
}

// ------------------------------------------- depthwise causal conv (k=4) + SiLU
__global__ void conv_silu_kernel(const float* __restrict__ xz, const float* __restrict__ w,
                                 const float* __restrict__ cb, float* __restrict__ xi) {
  const int idx = blockIdx.x * blockDim.x + threadIdx.x;
  if (idx >= NTOK * DINNER) return;
  const int d   = idx & (DINNER - 1);
  const int tok = idx >> 11;
  const int t   = tok & (LSEQ - 1);
  float acc = cb[d];
#pragma unroll
  for (int j = 0; j < DCONV; ++j) {
    const int tj = t - (DCONV - 1) + j;
    if (tj >= 0)
      acc += w[d * DCONV + j] * xz[(size_t)(tok - (DCONV - 1) + j) * DXZ + d];
  }
  xi[idx] = silu_f(acc);
}

// ------------------------------------------------ x_dbc = xi @ W_x (N=33, skinny)
__global__ void xdbc_kernel(const float* __restrict__ xi, const float* __restrict__ Wx,
                            float* __restrict__ xdbc) {
  __shared__ float sxi[DINNER];
  const int tok = blockIdx.x;
  for (int i = threadIdx.x; i < DINNER; i += blockDim.x)
    sxi[i] = xi[(size_t)tok * DINNER + i];
  __syncthreads();
  const int lane = threadIdx.x & 31, wave = threadIdx.x >> 5;   // 4 waves
  for (int j = wave; j < NDBC; j += 4) {
    float s = 0.f;
    for (int k = lane; k < DINNER; k += 32) s += sxi[k] * Wx[(size_t)k * NDBC + j];
    for (int off = 16; off > 0; off >>= 1) s += __shfl_down(s, off, 32);
    if (lane == 0) xdbc[(size_t)tok * NDBC + j] = s;
  }
}

// ---------------- chunked parallel scan over L: 3 phases, layouts [b][c][n][d]
// Phase A: per 64-step chunk, per channel: local state from h=0; emit (prod dA, h_end).
__global__ void scan_phaseA(const float* __restrict__ xdbc, const float* __restrict__ xi,
                            const float* __restrict__ Wdt, const float* __restrict__ bdt,
                            const float* __restrict__ logA,
                            float* __restrict__ Aprod, float* __restrict__ hEnd) {
  __shared__ float sbc[TCHUNK * NDBC];
  const int b = blockIdx.z, c = blockIdx.y;
  const int d = (blockIdx.x << 8) + threadIdx.x;
  float A[DSTATE], h[DSTATE], p[DSTATE];
#pragma unroll
  for (int n = 0; n < DSTATE; ++n) { A[n] = -__expf(logA[d * DSTATE + n]); h[n] = 0.f; p[n] = 1.f; }
  const float wdt = Wdt[d], bd = bdt[d];
  const int t0 = c * TCHUNK;
  for (int i = threadIdx.x; i < TCHUNK * NDBC; i += blockDim.x)
    sbc[i] = xdbc[(size_t)(b * LSEQ + t0) * NDBC + i];
  __syncthreads();
  for (int tt = 0; tt < TCHUNK; ++tt) {
    const float* row = &sbc[tt * NDBC];
    const float delta = softplus_f(row[0] * wdt + bd);
    const float dxv   = delta * xi[(size_t)(b * LSEQ + t0 + tt) * DINNER + d];
#pragma unroll
    for (int n = 0; n < DSTATE; ++n) {
      const float dA = __expf(delta * A[n]);
      p[n] *= dA;
      h[n]  = h[n] * dA + dxv * row[1 + n];
    }
  }
#pragma unroll
  for (int n = 0; n < DSTATE; ++n) {
    const size_t o = (((size_t)(b * NCHUNK + c) * DSTATE + n) * DINNER) + d;
    Aprod[o] = p[n];
    hEnd[o]  = h[n];
  }
}

// Phase B: serial prefix over the 16 chunk summaries, per (b,n,d) — fully coalesced in d.
__global__ void scan_phaseB(const float* __restrict__ Aprod, const float* __restrict__ hEnd,
                            float* __restrict__ hStart) {
  const int idx = blockIdx.x * blockDim.x + threadIdx.x;   // B*DSTATE*DINNER = 65536
  const int d = idx & (DINNER - 1);
  const int n = (idx >> 11) & (DSTATE - 1);
  const int b = idx >> 15;
  float h = 0.f;
  for (int c = 0; c < NCHUNK; ++c) {
    const size_t o = (((size_t)(b * NCHUNK + c) * DSTATE + n) * DINNER) + d;
    hStart[o] = h;
    h = Aprod[o] * h + hEnd[o];
  }
}

// Phase C: redo chunk scan seeded with correct h_start; emit y_t = C_t . h_t.
__global__ void scan_phaseC(const float* __restrict__ xdbc, const float* __restrict__ xi,
                            const float* __restrict__ Wdt, const float* __restrict__ bdt,
                            const float* __restrict__ logA, const float* __restrict__ hStart,
                            float* __restrict__ y) {
  __shared__ float sbc[TCHUNK * NDBC];
  const int b = blockIdx.z, c = blockIdx.y;
  const int d = (blockIdx.x << 8) + threadIdx.x;
  float A[DSTATE], h[DSTATE];
#pragma unroll
  for (int n = 0; n < DSTATE; ++n) {
    A[n] = -__expf(logA[d * DSTATE + n]);
    h[n] = hStart[(((size_t)(b * NCHUNK + c) * DSTATE + n) * DINNER) + d];
  }
  const float wdt = Wdt[d], bd = bdt[d];
  const int t0 = c * TCHUNK;
  for (int i = threadIdx.x; i < TCHUNK * NDBC; i += blockDim.x)
    sbc[i] = xdbc[(size_t)(b * LSEQ + t0) * NDBC + i];
  __syncthreads();
  for (int tt = 0; tt < TCHUNK; ++tt) {
    const float* row = &sbc[tt * NDBC];
    if (tt + 1 < TCHUNK)
      __builtin_prefetch(&xi[(size_t)(b * LSEQ + t0 + tt + 1) * DINNER + d], 0, 1);
    const float delta = softplus_f(row[0] * wdt + bd);
    const float dxv   = delta * xi[(size_t)(b * LSEQ + t0 + tt) * DINNER + d];
    float accy = 0.f;
#pragma unroll
    for (int n = 0; n < DSTATE; ++n) {
      const float dA = __expf(delta * A[n]);
      h[n] = h[n] * dA + dxv * row[1 + n];
      accy += h[n] * row[1 + DSTATE + n];
    }
    y[(size_t)(b * LSEQ + t0 + tt) * DINNER + d] = accy;
  }
}

// ------------------------------------------------------------- y2 = y * silu(z) -> bf16
__global__ void ymul_kernel(const float* __restrict__ y, const float* __restrict__ xz,
                            __bf16* __restrict__ y2) {
  const int idx = blockIdx.x * blockDim.x + threadIdx.x;
  if (idx >= NTOK * DINNER) return;
  const int d   = idx & (DINNER - 1);
  const int tok = idx >> 11;
  const float z = xz[(size_t)tok * DXZ + DINNER + d];
  y2[idx] = (__bf16)(y[idx] * silu_f(z));
}

// ----------------------------------------------------------------- launcher
extern "C" void kernel_launch(void* const* d_in, const int* in_sizes, int n_in,
                              void* d_out, int out_size, void* d_ws, size_t ws_size,
                              hipStream_t stream) {
  const float* x     = (const float*)d_in[0];
  const float* W_in  = (const float*)d_in[1];
  const float* convw = (const float*)d_in[2];
  const float* convb = (const float*)d_in[3];
  const float* W_x   = (const float*)d_in[4];
  const float* W_dt  = (const float*)d_in[5];
  const float* b_dt  = (const float*)d_in[6];
  const float* W_out = (const float*)d_in[7];
  const float* logA  = (const float*)d_in[8];
  const float* ln_g  = (const float*)d_in[9];
  const float* ln_b  = (const float*)d_in[10];
  float* out = (float*)d_out;

  char* p = (char*)d_ws;
  auto carve = [&](size_t bytes) { char* q = p; p += (bytes + 255) & ~(size_t)255; return q; };
  __bf16* xnb   = (__bf16*)carve((size_t)NTOK * DMODEL * 2);        // 4 MB
  __bf16* WtIn  = (__bf16*)carve((size_t)DXZ * DMODEL * 2);         // 8 MB  [4096][1024]
  __bf16* WtOut = (__bf16*)carve((size_t)DMODEL * DINNER * 2);      // 4 MB  [1024][2048]
  float*  xz    = (float*) carve((size_t)NTOK * DXZ * 4);           // 32 MB
  float*  xi    = (float*) carve((size_t)NTOK * DINNER * 4);        // 16 MB
  float*  xdbc  = (float*) carve((size_t)NTOK * NDBC * 4);          // 0.27 MB
  float*  y     = (float*) carve((size_t)NTOK * DINNER * 4);        // 16 MB
  __bf16* y2    = (__bf16*)carve((size_t)NTOK * DINNER * 2);        // 8 MB
  float*  Aprod = (float*) carve((size_t)B_SZ * NCHUNK * DSTATE * DINNER * 4); // 4 MB
  float*  hEnd  = (float*) carve((size_t)B_SZ * NCHUNK * DSTATE * DINNER * 4); // 4 MB
  float*  hStrt = (float*) carve((size_t)B_SZ * NCHUNK * DSTATE * DINNER * 4); // 4 MB

  // one-time weight transpose+convert (bf16 [N][K])
  transpose_bf16_kernel<<<dim3(DXZ / 32, DMODEL / 32), dim3(32, 8), 0, stream>>>(W_in, WtIn, DMODEL, DXZ);
  transpose_bf16_kernel<<<dim3(DMODEL / 32, DINNER / 32), dim3(32, 8), 0, stream>>>(W_out, WtOut, DINNER, DMODEL);

  ln_kernel<<<NTOK, 256, 0, stream>>>(x, ln_g, ln_b, xnb);

  wmma_gemm_bf16<false><<<dim3(DXZ / 128, NTOK / 128), 256, 0, stream>>>(
      xnb, WtIn, nullptr, xz, NTOK, DXZ, DMODEL);

  const int ne = NTOK * DINNER;
  conv_silu_kernel<<<(ne + 255) / 256, 256, 0, stream>>>(xz, convw, convb, xi);

  xdbc_kernel<<<NTOK, 128, 0, stream>>>(xi, W_x, xdbc);

  const dim3 sgrid(DINNER / 256, NCHUNK, B_SZ);   // 8 x 16 x 2 = 256 blocks
  scan_phaseA<<<sgrid, 256, 0, stream>>>(xdbc, xi, W_dt, b_dt, logA, Aprod, hEnd);
  scan_phaseB<<<(B_SZ * DSTATE * DINNER) / 256, 256, 0, stream>>>(Aprod, hEnd, hStrt);
  scan_phaseC<<<sgrid, 256, 0, stream>>>(xdbc, xi, W_dt, b_dt, logA, hStrt, y);

  ymul_kernel<<<(ne + 255) / 256, 256, 0, stream>>>(y, xz, y2);

  wmma_gemm_bf16<true><<<dim3(DMODEL / 128, NTOK / 128), 256, 0, stream>>>(
      y2, WtOut, x, out, NTOK, DMODEL, DINNER);
}
